// FrequencyFeatureExtractor_72507637891532
// MI455X (gfx1250) — matrix operand order repaired and checked by reading, not compiled, not run on hardware
//
#include <hip/hip_runtime.h>

// ---------------------------------------------------------------------------
// Problem constants
// ---------------------------------------------------------------------------
#define BB    128          // batch
#define IMG   65536        // 256*256
#define NSIDE 256
#define FEAT  512
#define NBLK  63
#define NBINS 2048

typedef _Float16 h16;
typedef __attribute__((ext_vector_type(4)))  _Float16 v4h;
typedef __attribute__((ext_vector_type(8)))  _Float16 v8h;
typedef __attribute__((ext_vector_type(16))) _Float16 v16h;
typedef __attribute__((ext_vector_type(8)))  float    v8f;

union H16 { v16h v; v8h h[2]; };

// ---------------------------------------------------------------------------
// WMMA helpers (CDNA5 V_WMMA_F32_16X16X32_F16, wave32)
//
// A (16x32 f16): lane L holds row M = L&15; elems 0..7 -> K = kb..kb+7,
//                elems 8..15 -> K = kb+16..kb+23, kb = 8*(L>>4).
// B (32x16 f16): lane L holds col N = L&15; elems 0..7 -> K = kb..kb+7,
//                elems 8..15 -> K = kb+16..kb+23, kb = 8*(L>>4).
// C/D (16x16 f32): VGPR r: lanes 0-15 -> (M=r, N=lane),
//                          lanes 16-31 -> (M=r+8, N=lane-16).
// ---------------------------------------------------------------------------
__device__ __forceinline__ v8f wmma_acc(v16h a, v16h b, v8f c) {
  return __builtin_amdgcn_wmma_f32_16x16x32_f16(
      /*neg_a=*/false, a, /*neg_b=*/false, b,
      /*c_mod=*/(short)0, c, /*reuse_a=*/false, /*reuse_b=*/false);
}

__device__ __forceinline__ v16h loadA16x32(const h16* base, int row0, int lda, int k0) {
  const int lane = threadIdx.x & 31;
  const int row  = row0 + (lane & 15);
  const int kb   = (lane >> 4) << 3;                 // 0 or 8
  const h16* p = base + (size_t)row * lda + k0 + kb; // 16B aligned
  H16 r;
  r.h[0] = *(const v8h*)(p);        // K = kb .. kb+7
  r.h[1] = *(const v8h*)(p + 16);   // K = kb+16 .. kb+23
  return r.v;
}

// Reads B(K,N) = base[(col0+N)*lda + k0+K]  (i.e. the operand matrix stored
// N-major; for symmetric DFT matrices this equals the K-major matrix).
__device__ __forceinline__ v16h loadB32x16(const h16* base, int col0, int lda, int k0) {
  const int lane = threadIdx.x & 31;
  const int col  = col0 + (lane & 15);
  const int kb   = (lane >> 4) << 3;                 // 0 or 8
  const h16* p = base + (size_t)col * lda + k0 + kb;
  H16 r;
  r.h[0] = *(const v8h*)(p);
  r.h[1] = *(const v8h*)(p + 16);
  return r.v;
}

// ---------------------------------------------------------------------------
// 1) grayscale:  gray = .299 R + .587 G + .114 B   (f32 -> f16)
// ---------------------------------------------------------------------------
__global__ void k_gray(const float* __restrict__ x, h16* __restrict__ grayH) {
  const size_t i = (size_t)blockIdx.x * blockDim.x + threadIdx.x;
  const size_t base = i * 4;
  if (base >= (size_t)BB * IMG) return;
  const size_t b = base >> 16;
  const size_t p = base & (IMG - 1);
  const float4 r  = *(const float4*)(x + (b * 3 + 0) * IMG + p);
  const float4 g  = *(const float4*)(x + (b * 3 + 1) * IMG + p);
  const float4 bl = *(const float4*)(x + (b * 3 + 2) * IMG + p);
  v4h o;
  o[0] = (h16)(0.299f * r.x + 0.587f * g.x + 0.114f * bl.x);
  o[1] = (h16)(0.299f * r.y + 0.587f * g.y + 0.114f * bl.y);
  o[2] = (h16)(0.299f * r.z + 0.587f * g.z + 0.114f * bl.z);
  o[3] = (h16)(0.299f * r.w + 0.587f * g.w + 0.114f * bl.w);
  *(v4h*)(grayH + base) = o;
}

// ---------------------------------------------------------------------------
// 2) DFT matrices (symmetric): C[k,n]=cos(2pi kn/256), Sn=-sin, Sp=+sin
// ---------------------------------------------------------------------------
__global__ void k_dftmats(h16* __restrict__ dC, h16* __restrict__ dSn,
                          h16* __restrict__ dSp) {
  const int i = blockIdx.x * blockDim.x + threadIdx.x;
  if (i >= IMG) return;
  const int k = i >> 8, n = i & 255;
  const int t = (k * n) & 255;                        // periodic reduction
  const float ang = -0.0245436926f * (float)t;        // -2*pi/256 * t
  float s, c;
  __sincosf(ang, &s, &c);
  dC[i]  = (h16)c;
  dSn[i] = (h16)s;     // = -sin(2pi kn/256)
  dSp[i] = (h16)(-s);  // = +sin(2pi kn/256)
}

// ---------------------------------------------------------------------------
// 3) Row DFT:  S = G @ F   (Sr = G@C, Si = G@Sn), result stored TRANSPOSED
//    so the column pass gets contiguous B fragments.
//    One wave per 16x16 tile; 4 waves/block; grid (64, B).
// ---------------------------------------------------------------------------
__global__ void k_dft_pass1(const h16* __restrict__ gray, const h16* __restrict__ dC,
                            const h16* __restrict__ dSn,
                            h16* __restrict__ SrT, h16* __restrict__ SiT) {
  const int b    = blockIdx.y;
  const int wave = threadIdx.x >> 5;
  const int tile = blockIdx.x * 4 + wave;     // 0..255
  const int rt = tile >> 4, ct = tile & 15;
  const h16* G = gray + (size_t)b * IMG;
  v8f accR = {}; v8f accI = {};
  for (int k0 = 0; k0 < 256; k0 += 32) {
    if (k0 + 32 < 256) __builtin_prefetch((const void*)(G + (size_t)(rt * 16) * 256 + k0 + 32), 0, 1);
    const v16h a  = loadA16x32(G,   rt * 16, 256, k0);
    const v16h bc = loadB32x16(dC,  ct * 16, 256, k0);   // C symmetric
    const v16h bs = loadB32x16(dSn, ct * 16, 256, k0);   // Sn symmetric
    accR = wmma_acc(a, bc, accR);
    accI = wmma_acc(a, bs, accI);
  }
  const int lane  = threadIdx.x & 31;
  const int col   = ct * 16 + (lane & 15);
  const int rbase = rt * 16 + ((lane >> 4) << 3);
  h16* Sr = SrT + (size_t)b * IMG;
  h16* Si = SiT + (size_t)b * IMG;
#pragma unroll
  for (int r = 0; r < 8; ++r) {
    const int row = rbase + r;
    Sr[(size_t)col * 256 + row] = (h16)accR[r];   // transposed store
    Si[(size_t)col * 256 + row] = (h16)accI[r];
  }
}

// ---------------------------------------------------------------------------
// 4) Column DFT + magnitude + fftshift:
//    Mr = C@Sr + sin@Si ;  Mi = C@Si + (-sin)@Sr ;  mag = sqrt(Mr^2+Mi^2)
// ---------------------------------------------------------------------------
__global__ void k_dft_pass2(const h16* __restrict__ dC, const h16* __restrict__ dSp,
                            const h16* __restrict__ dSn,
                            const h16* __restrict__ SrT, const h16* __restrict__ SiT,
                            float* __restrict__ mag) {
  const int b    = blockIdx.y;
  const int wave = threadIdx.x >> 5;
  const int tile = blockIdx.x * 4 + wave;
  const int rt = tile >> 4, ct = tile & 15;
  const h16* Sr = SrT + (size_t)b * IMG;
  const h16* Si = SiT + (size_t)b * IMG;
  v8f accR = {}; v8f accI = {};
  for (int k0 = 0; k0 < 256; k0 += 32) {
    if (k0 + 32 < 256) __builtin_prefetch((const void*)(Sr + (size_t)(ct * 16) * 256 + k0 + 32), 0, 1);
    const v16h aC  = loadA16x32(dC,  rt * 16, 256, k0);
    const v16h aSp = loadA16x32(dSp, rt * 16, 256, k0);
    const v16h aSn = loadA16x32(dSn, rt * 16, 256, k0);
    const v16h bR  = loadB32x16(Sr,  ct * 16, 256, k0);  // SrT is N-major
    const v16h bI  = loadB32x16(Si,  ct * 16, 256, k0);
    accR = wmma_acc(aC,  bR, accR);
    accR = wmma_acc(aSp, bI, accR);
    accI = wmma_acc(aC,  bI, accI);
    accI = wmma_acc(aSn, bR, accI);
  }
  const int lane  = threadIdx.x & 31;
  const int col   = ct * 16 + (lane & 15);
  const int rbase = rt * 16 + ((lane >> 4) << 3);
  float* M = mag + (size_t)b * IMG;
#pragma unroll
  for (int r = 0; r < 8; ++r) {
    const int row = rbase + r;
    const float v = sqrtf(accR[r] * accR[r] + accI[r] * accI[r]);
    const int sy = (row + 128) & 255;                 // fftshift
    const int sx = (col + 128) & 255;
    M[(size_t)sy * 256 + sx] = v;
  }
}

// ---------------------------------------------------------------------------
// 5) FFT statistics: one 1024-thread workgroup per image.
//    Pass A: moments/max per region; Pass B: 2048-bin histograms + hv count;
//    percentiles by interpolated rank selection in the histogram.
// ---------------------------------------------------------------------------
__device__ __forceinline__ int binof(float v, float sc) {
  int t = (int)(v * sc);
  return t > (NBINS - 1) ? (NBINS - 1) : (t < 0 ? 0 : t);
}

__device__ float pctile(const unsigned* h, float n, float maxv, float q) {
  const float t = q * (n - 1.0f) + 1.0f;   // target cumulative count
  float cum = 0.f;
  for (int i = 0; i < NBINS; ++i) {
    const float c = (float)h[i];
    if (c > 0.f && cum + c >= t) {
      const float frac = (t - cum) / c;
      return (((float)i + frac) * maxv) / (float)NBINS;
    }
    cum += c;
  }
  return maxv;
}

__global__ __launch_bounds__(1024) void k_fft_stats(const float* __restrict__ mag,
                                                    float* __restrict__ feat) {
  const int b   = blockIdx.x;
  const int tid = threadIdx.x;
  const float* M = mag + (size_t)b * IMG;

  __shared__ float    s_sum[5], s_sq[5];
  __shared__ unsigned s_max[4];
  __shared__ unsigned s_cnt[5];
  __shared__ float    s_line[4];
  __shared__ unsigned s_hist[4][NBINS];
  __shared__ unsigned s_hvcnt;
  __shared__ float    s_feat[24];

  if (tid < 5) { s_sum[tid] = 0.f; s_sq[tid] = 0.f; s_cnt[tid] = 0u; }
  if (tid < 4) { s_max[tid] = 0u; s_line[tid] = 0.f; }
  if (tid == 0) s_hvcnt = 0u;
  for (int i = tid; i < 4 * NBINS; i += 1024) (&s_hist[0][0])[i] = 0u;
  __syncthreads();

  float sm0=0,sq0=0,mx0=0; unsigned c0=0;
  float sm1=0,sq1=0,mx1=0; unsigned c1=0;
  float sm2=0,sq2=0,mx2=0; unsigned c2=0;
  float sm3=0,sq3=0,mx3=0;
  float sm4=0,sq4=0;       unsigned c4=0;
  float hls=0,hlq=0,vls=0,vlq=0;
  for (int i = tid; i < IMG; i += 1024) {
    const float v = M[i];
    const int y = i >> 8, x = i & 255;
    const int dx = x - 128, dy = y - 128;
    const int d2 = dx * dx + dy * dy;
    if (d2 < 400)        { sm0 += v; sq0 += v*v; mx0 = fmaxf(mx0, v); c0++; }
    else if (d2 < 2500)  { sm1 += v; sq1 += v*v; mx1 = fmaxf(mx1, v); c1++; }
    else if (d2 < 10000) { sm2 += v; sq2 += v*v; mx2 = fmaxf(mx2, v); c2++; }
    sm3 += v; sq3 += v*v; mx3 = fmaxf(mx3, v);
    if (d2 > 6400) { sm4 += v; sq4 += v*v; c4++; }
    if (y == 128)  { hls += v; hlq += v*v; }
    if (x == 128)  { vls += v; vlq += v*v; }
  }
  atomicAdd(&s_sum[0], sm0); atomicAdd(&s_sq[0], sq0); atomicMax(&s_max[0], __float_as_uint(mx0)); atomicAdd(&s_cnt[0], c0);
  atomicAdd(&s_sum[1], sm1); atomicAdd(&s_sq[1], sq1); atomicMax(&s_max[1], __float_as_uint(mx1)); atomicAdd(&s_cnt[1], c1);
  atomicAdd(&s_sum[2], sm2); atomicAdd(&s_sq[2], sq2); atomicMax(&s_max[2], __float_as_uint(mx2)); atomicAdd(&s_cnt[2], c2);
  atomicAdd(&s_sum[3], sm3); atomicAdd(&s_sq[3], sq3); atomicMax(&s_max[3], __float_as_uint(mx3));
  atomicAdd(&s_sum[4], sm4); atomicAdd(&s_sq[4], sq4); atomicAdd(&s_cnt[4], c4);
  atomicAdd(&s_line[0], hls); atomicAdd(&s_line[1], hlq);
  atomicAdd(&s_line[2], vls); atomicAdd(&s_line[3], vlq);
  __syncthreads();

  const float m0 = __uint_as_float(s_max[0]), sc0 = m0 > 0.f ? (float)NBINS / m0 : 0.f;
  const float m1 = __uint_as_float(s_max[1]), sc1 = m1 > 0.f ? (float)NBINS / m1 : 0.f;
  const float m2 = __uint_as_float(s_max[2]), sc2 = m2 > 0.f ? (float)NBINS / m2 : 0.f;
  const float m3 = __uint_as_float(s_max[3]), sc3 = m3 > 0.f ? (float)NBINS / m3 : 0.f;
  const float hvthr = 2.f * s_sum[4] / fmaxf((float)s_cnt[4], 1.f);
  unsigned hvloc = 0;
  for (int i = tid; i < IMG; i += 1024) {
    const float v = M[i];
    const int y = i >> 8, x = i & 255;
    const int dx = x - 128, dy = y - 128;
    const int d2 = dx * dx + dy * dy;
    if (d2 < 400)        atomicAdd(&s_hist[0][binof(v, sc0)], 1u);
    else if (d2 < 2500)  atomicAdd(&s_hist[1][binof(v, sc1)], 1u);
    else if (d2 < 10000) atomicAdd(&s_hist[2][binof(v, sc2)], 1u);
    atomicAdd(&s_hist[3][binof(v, sc3)], 1u);
    if (d2 > 6400 && v > hvthr) hvloc++;
  }
  atomicAdd(&s_hvcnt, hvloc);
  __syncthreads();

  if (tid == 0) {
    float mean, sd;
    // bands
    mean = s_sum[0] / (float)s_cnt[0];
    sd = sqrtf(fmaxf(0.f, s_sq[0] / (float)s_cnt[0] - mean * mean));
    s_feat[0] = mean; s_feat[1] = sd; s_feat[2] = m0;
    s_feat[3] = pctile(s_hist[0], (float)s_cnt[0], m0, 0.95f);
    mean = s_sum[1] / (float)s_cnt[1];
    sd = sqrtf(fmaxf(0.f, s_sq[1] / (float)s_cnt[1] - mean * mean));
    s_feat[4] = mean; s_feat[5] = sd; s_feat[6] = m1;
    s_feat[7] = pctile(s_hist[1], (float)s_cnt[1], m1, 0.95f);
    mean = s_sum[2] / (float)s_cnt[2];
    sd = sqrtf(fmaxf(0.f, s_sq[2] / (float)s_cnt[2] - mean * mean));
    s_feat[8] = mean; s_feat[9] = sd; s_feat[10] = m2;
    s_feat[11] = pctile(s_hist[2], (float)s_cnt[2], m2, 0.95f);
    // flat
    mean = s_sum[3] / (float)IMG;
    sd = sqrtf(fmaxf(0.f, s_sq[3] / (float)IMG - mean * mean));
    s_feat[12] = mean; s_feat[13] = sd; s_feat[14] = m3;
    s_feat[15] = pctile(s_hist[3], (float)IMG, m3, 0.95f);
    s_feat[16] = pctile(s_hist[3], (float)IMG, m3, 0.05f);
    // center lines
    mean = s_line[0] / 256.f;
    s_feat[17] = mean;
    s_feat[18] = sqrtf(fmaxf(0.f, s_line[1] / 256.f - mean * mean));
    mean = s_line[2] / 256.f;
    s_feat[19] = mean;
    s_feat[20] = sqrtf(fmaxf(0.f, s_line[3] / 256.f - mean * mean));
    // high-frequency region
    const float hm = s_sum[4] / fmaxf((float)s_cnt[4], 1.f);
    s_feat[21] = hm;
    s_feat[22] = sqrtf(fmaxf(0.f, s_sq[4] / fmaxf((float)s_cnt[4], 1.f) - hm * hm));
    s_feat[23] = (float)s_hvcnt;
  }
  __syncthreads();
  float* F = feat + (size_t)b * 256;
  if (tid < 24)       F[tid] = s_feat[tid];
  else if (tid < 256) F[tid] = 0.f;
}

// ---------------------------------------------------------------------------
// 6) DCT block features: one 64-thread block per (image, 8x8 block)
// ---------------------------------------------------------------------------
__global__ void k_dct(const h16* __restrict__ gray, float* __restrict__ dct_feat) {
  const int n = blockIdx.x;   // block index 0..62
  const int b = blockIdx.y;
  const int tid = threadIdx.x;
  const int i = tid >> 3, j = tid & 7;
  __shared__ float s_blk[8][8];
  __shared__ float s_D[8][8];
  __shared__ float s_dct[64];
  const int by = n >> 5, bx = n & 31;
  s_blk[i][j] = (float)gray[(size_t)b * IMG + (by * 8 + i) * 256 + bx * 8 + j];
  const float cc = (i == 0) ? 0.3535533906f : 0.5f;   // sqrt(2/8) * (1/sqrt2 for row 0)
  s_D[i][j] = cc * cosf(3.14159265358979f * (float)(2 * j + 1) * (float)i / 16.0f);
  __syncthreads();
  float acc = 0.f;
  const int l = j;
#pragma unroll
  for (int jj = 0; jj < 8; ++jj) {
    float t = 0.f;
#pragma unroll
    for (int k = 0; k < 8; ++k) t += s_blk[jj][k] * s_D[l][k];
    acc += s_D[i][jj] * t;
  }
  s_dct[i * 8 + l] = acc;
  __syncthreads();
  if (tid == 0) {
    float sum = 0.f, sq = 0.f, asum = 0.f, amax = 0.f;
    for (int m = 1; m < 64; ++m) {
      const float v = s_dct[m];
      sum += v; sq += v * v;
      const float av = fabsf(v);
      asum += av; amax = fmaxf(amax, av);
    }
    const float mean = sum / 63.f;
    const float sd = sqrtf(fmaxf(0.f, sq / 63.f - mean * mean));
    int cnt = 0;
    for (int m = 1; m < 64; ++m) if (fabsf(s_dct[m]) > sd) cnt++;
    float* F = dct_feat + (size_t)b * 256 + n * 4;
    F[0] = asum / 63.f; F[1] = sd; F[2] = amax; F[3] = (float)cnt;
    if (n == 0) {
      float* P = dct_feat + (size_t)b * 256;
      P[252] = 0.f; P[253] = 0.f; P[254] = 0.f; P[255] = 0.f;
    }
  }
}

// ---------------------------------------------------------------------------
// 7) f32 -> f16 pack (features and weights) and WMMA feature heads
// ---------------------------------------------------------------------------
__global__ void k_f2h(const float* __restrict__ src, h16* __restrict__ dst, int n) {
  const int i = blockIdx.x * blockDim.x + threadIdx.x;
  if (i < n) dst[i] = (h16)src[i];
}

// out[m, off+n] = sum_k feat[m,k] * W[n,k] + bias[n]   (M=128,N=512,K=256)
__global__ void k_head(const h16* __restrict__ featH, const h16* __restrict__ Wh,
                       const float* __restrict__ bias, float* __restrict__ out, int off) {
  const int wave = threadIdx.x >> 5;
  const int tile = blockIdx.x * 4 + wave;   // 0..255
  const int rt = tile >> 5;                 // 8 M-tiles
  const int ct = tile & 31;                 // 32 N-tiles
  v8f acc = {};
  for (int k0 = 0; k0 < 256; k0 += 32) {
    const v16h a = loadA16x32(featH, rt * 16, 256, k0);
    const v16h w = loadB32x16(Wh,    ct * 16, 256, k0);  // W is N-major (512x256)
    acc = wmma_acc(a, w, acc);
  }
  const int lane  = threadIdx.x & 31;
  const int col   = ct * 16 + (lane & 15);
  const int rbase = rt * 16 + ((lane >> 4) << 3);
#pragma unroll
  for (int r = 0; r < 8; ++r) {
    const int row = rbase + r;
    out[(size_t)row * 1024 + off + col] = acc[r] + bias[col];
  }
}

// ---------------------------------------------------------------------------
// Launch
// ---------------------------------------------------------------------------
extern "C" void kernel_launch(void* const* d_in, const int* in_sizes, int n_in,
                              void* d_out, int out_size, void* d_ws, size_t ws_size,
                              hipStream_t stream) {
  (void)in_sizes; (void)n_in; (void)out_size; (void)ws_size;
  const float* x  = (const float*)d_in[0];
  const float* Wf = (const float*)d_in[1];
  const float* bf = (const float*)d_in[2];
  const float* Wd = (const float*)d_in[3];
  const float* bd = (const float*)d_in[4];
  float* out = (float*)d_out;

  char* ws = (char*)d_ws;
  size_t off = 0;
  auto take = [&](size_t bytes) -> void* {
    void* p = ws + off;
    off = (off + bytes + 255) & ~(size_t)255;
    return p;
  };
  h16*   grayH = (h16*)  take((size_t)BB * IMG * 2);   // 16 MB
  h16*   dC    = (h16*)  take((size_t)IMG * 2);
  h16*   dSn   = (h16*)  take((size_t)IMG * 2);
  h16*   dSp   = (h16*)  take((size_t)IMG * 2);
  h16*   SrT   = (h16*)  take((size_t)BB * IMG * 2);   // 16 MB
  h16*   SiT   = (h16*)  take((size_t)BB * IMG * 2);   // 16 MB
  float* mag   = (float*)take((size_t)BB * IMG * 4);   // 32 MB
  float* ff    = (float*)take((size_t)BB * 256 * 4);
  float* df    = (float*)take((size_t)BB * 256 * 4);
  h16*   ffh   = (h16*)  take((size_t)BB * 256 * 2);
  h16*   dfh   = (h16*)  take((size_t)BB * 256 * 2);
  h16*   WfH   = (h16*)  take((size_t)FEAT * 256 * 2);
  h16*   WdH   = (h16*)  take((size_t)FEAT * 256 * 2);

  k_gray   <<<dim3((BB * IMG / 4) / 256), 256, 0, stream>>>(x, grayH);
  k_dftmats<<<dim3(IMG / 256),            256, 0, stream>>>(dC, dSn, dSp);

  k_dft_pass1<<<dim3(64, BB), 128, 0, stream>>>(grayH, dC, dSn, SrT, SiT);
  k_dft_pass2<<<dim3(64, BB), 128, 0, stream>>>(dC, dSp, dSn, SrT, SiT, mag);

  k_fft_stats<<<dim3(BB), 1024, 0, stream>>>(mag, ff);
  k_dct      <<<dim3(NBLK, BB), 64, 0, stream>>>(grayH, df);

  k_f2h<<<dim3((BB * 256 + 255) / 256),   256, 0, stream>>>(ff, ffh, BB * 256);
  k_f2h<<<dim3((BB * 256 + 255) / 256),   256, 0, stream>>>(df, dfh, BB * 256);
  k_f2h<<<dim3((FEAT * 256 + 255) / 256), 256, 0, stream>>>(Wf, WfH, FEAT * 256);
  k_f2h<<<dim3((FEAT * 256 + 255) / 256), 256, 0, stream>>>(Wd, WdH, FEAT * 256);

  k_head<<<dim3(64), 128, 0, stream>>>(ffh, WfH, bf, out, 0);
  k_head<<<dim3(64), 128, 0, stream>>>(dfh, WdH, bd, out, FEAT);
}
